// WindowAttention_22230750724640
// MI455X (gfx1250) — matrix operand order repaired and checked by reading, not compiled, not run on hardware
//
#include <hip/hip_runtime.h>
#include <cstdint>

// ---------------------------------------------------------------------------
// Window attention (Swin-style) for MI455X / gfx1250, wave32 + WMMA bf16.
// One workgroup (8 wave32) per 8x8 window; fully fused QKV -> softmax(QK^T)V
// -> proj, everything on-chip in LDS (154KB -> 2 WGs/WGP).
// x window staged via global_load_async_to_lds_b128 (ASYNCcnt path), weights
// pre-converted to bf16 [N][K] in d_ws (L2-resident) and L2-prefetched.
// ---------------------------------------------------------------------------

#define CDIM   256
#define HEADS  8
#define HD     32
#define NTOK   64            // tokens per 8x8 window
#define XLD    264           // padded bf16 stride for 256-wide rows
#define QLD    34            // padded bf16 stride for 32-wide (head-dim) rows
#define VLD    66            // padded bf16 stride for 64-wide (token) rows
#define ATTN_SCALE 0.17677669529663687f   // 32^-0.5

typedef __attribute__((ext_vector_type(16))) __bf16 v16bf;
typedef __attribute__((ext_vector_type(8)))  float  v8f;

union Frag { v16bf v; unsigned int u[8]; };

__device__ __forceinline__ unsigned short f32_to_bf16(float f) {
  unsigned int u = __float_as_uint(f);
  u += 0x7FFFu + ((u >> 16) & 1u);           // round-to-nearest-even
  return (unsigned short)(u >> 16);
}

// A fragment (16x32 bf16): storage row-major [m][k], stride ld (bf16 elems).
__device__ __forceinline__ void load_a(const unsigned short* base, int ld,
                                       int row0, int k0, int lane, Frag& f) {
  const int m = lane & 15, half = lane >> 4;
  const unsigned short* p = base + (row0 + m) * ld + k0 + (half << 4);
#pragma unroll
  for (int j = 0; j < 4; ++j) f.u[j]     = *(const unsigned int*)(p + (j << 1));
#pragma unroll
  for (int j = 0; j < 4; ++j) f.u[4 + j] = *(const unsigned int*)(p + 8 + (j << 1));
}

// B fragment (32x16 bf16): storage [n][k] contiguous in k, stride ld.
__device__ __forceinline__ void load_b(const unsigned short* base, size_t ld,
                                       int col0, int k0, int lane, Frag& f) {
  const int n = lane & 15, half = lane >> 4;
  const unsigned short* p = base + (size_t)(col0 + n) * ld + k0 + (half << 4);
#pragma unroll
  for (int j = 0; j < 8; ++j) f.u[j] = *(const unsigned int*)(p + (j << 1));
}

__device__ __forceinline__ v8f wmma_bf16(const Frag& a, const Frag& b, v8f c) {
  return __builtin_amdgcn_wmma_f32_16x16x32_bf16(
      false, a.v, false, b.v, (short)0, c, false, false);
}

// ---------------------------------------------------------------------------
// Weight prep: f32 [K][N] -> bf16 [N][K] (transposed so B-frags are
// contiguous u32 loads). Runs once per launch; output lives in d_ws (L2).
// ---------------------------------------------------------------------------
__global__ void wa_convert_w(const float* __restrict__ w, unsigned short* __restrict__ wT,
                             int K, int N) {
  int idx = blockIdx.x * blockDim.x + threadIdx.x;
  if (idx < K * N) {
    int k = idx / N, n = idx % N;               // coalesced read
    wT[(size_t)n * K + k] = f32_to_bf16(w[idx]);
  }
}

// ---------------------------------------------------------------------------
// Fused window-attention kernel. grid = 4096 windows, block = 256 (8 waves).
// ---------------------------------------------------------------------------
__global__ __launch_bounds__(256)
void wa_window_attn(const float* __restrict__ x,
                    const unsigned short* __restrict__ wqkvT,   // [768][256] bf16
                    const float* __restrict__ bqkv,             // [768] f32
                    const unsigned short* __restrict__ wprojT,  // [256][256] bf16
                    const float* __restrict__ bproj,            // [256] f32
                    float* __restrict__ out) {
  // sQK doubles as the 64KB f32 async staging buffer during Phase A.
  __shared__ __align__(16) unsigned short sQK[2 * HEADS * NTOK * QLD]; // 69632 B
  __shared__ __align__(16) unsigned short sX [NTOK * XLD];             // 33792 B (aliased as O)
  __shared__ __align__(16) unsigned short sVt[HEADS * HD * VLD];       // 33792 B
  __shared__ __align__(16) unsigned short sP [8 * 16 * VLD];           // 16896 B
  unsigned short* sQ = sQK;
  unsigned short* sK = sQK + HEADS * NTOK * QLD;
  float* stage = (float*)sQK;                                          // [64][256] f32
#define sO sX

  const int tid  = threadIdx.x;
  // Force the wave id into an SGPR: makes tile indices, q/k/v selection and
  // all per-wave base addresses provably uniform (SALU + s_cbranch, no exec
  // divergence in the QKV scatter).
  const int wave = __builtin_amdgcn_readfirstlane(tid >> 5);
  const int lane = tid & 31;
  const int nloc = lane & 15, half = lane >> 4;

  const int wi  = blockIdx.x;
  const int b   = wi >> 10;
  const int nhi = (wi >> 5) & 31;
  const int nwi = wi & 31;

  // ---- Phase A1: async-copy x window (f32) directly into LDS staging ----
  // 64 rows x 64 16B-chunks; each thread issues 16 async b128 copies.
  {
    const unsigned stage_base = (unsigned)(uintptr_t)(void*)sQK; // LDS offset
    for (int idx = tid; idx < NTOK * 64; idx += 256) {
      int t = idx >> 6, cc = (idx & 63) << 2;
      int h = nhi * 8 + (t >> 3), w = nwi * 8 + (t & 7);
      const float* gp = x + (((size_t)(b * 256 + h)) * 256 + w) * 256 + cc;
      unsigned lds = stage_base + (unsigned)((t * 256 + cc) << 2);
      asm volatile("global_load_async_to_lds_b128 %0, %1, off"
                   :: "v"(lds), "v"((unsigned long long)(uintptr_t)gp)
                   : "memory");
    }
    // While the async copy is in flight, warm L2 with this wave's weight slice.
    const char* wq = (const char*)(wqkvT + (size_t)(wave * 6 * 16) * CDIM); // 48KB
#pragma unroll 4
    for (int p = lane; p < 384; p += 32) __builtin_prefetch(wq + p * 128, 0, 0);
    const char* wp = (const char*)(wprojT + (size_t)(wave * 2 * 16) * CDIM); // 16KB
#pragma unroll 4
    for (int p = lane; p < 128; p += 32) __builtin_prefetch(wp + p * 128, 0, 0);
    asm volatile("s_wait_asynccnt 0x0" ::: "memory");
  }
  __syncthreads();

  // ---- Phase A2: convert staged f32 -> bf16 X (LDS->LDS) ----
  for (int idx = tid; idx < NTOK * 64; idx += 256) {
    int t = idx >> 6, cc = (idx & 63) << 2;
    float4 v = *(const float4*)(stage + t * 256 + cc);
    unsigned lo = (unsigned)f32_to_bf16(v.x) | ((unsigned)f32_to_bf16(v.y) << 16);
    unsigned hi = (unsigned)f32_to_bf16(v.z) | ((unsigned)f32_to_bf16(v.w) << 16);
    *(unsigned int*)(sX + t * XLD + cc)     = lo;
    *(unsigned int*)(sX + t * XLD + cc + 2) = hi;
  }
  __syncthreads();   // staging dead; sQ/sK may now be written

  // ---- Phase B: QKV = X[64x256] @ Wqkv[256x768] + b; scatter to Q/K/V^T ----
  // Per wave: nt in [wave*6, wave*6+6), all 4 M-tiles; A-frags hoisted per mt.
  for (int mt = 0; mt < 4; ++mt) {
    Frag a[8];
#pragma unroll
    for (int ks = 0; ks < 8; ++ks) load_a(sX, XLD, mt * 16, ks * 32, lane, a[ks]);
    for (int ii = 0; ii < 6; ++ii) {
      const int nt = wave * 6 + ii;              // scalar
      v8f acc = {0.f, 0.f, 0.f, 0.f, 0.f, 0.f, 0.f, 0.f};
#pragma unroll
      for (int ks = 0; ks < 8; ++ks) {
        Frag bb;
        load_b(wqkvT, CDIM, nt * 16, ks * 32, lane, bb);
        acc = wmma_bf16(a[ks], bb, acc);
      }
      float bias = bqkv[nt * 16 + nloc];
      // part/head derived from scalar nt -> s_cbranch, no exec divergence
      const int part = nt >> 4;
      const int head = (nt >> 1) & 7;
      const int d    = ((nt & 1) << 4) + nloc;
      if (part == 0) {
#pragma unroll
        for (int j = 0; j < 8; ++j) {
          int t = mt * 16 + j + (half << 3);
          sQ[(head * NTOK + t) * QLD + d] = f32_to_bf16(acc[j] + bias);
        }
      } else if (part == 1) {
#pragma unroll
        for (int j = 0; j < 8; ++j) {
          int t = mt * 16 + j + (half << 3);
          sK[(head * NTOK + t) * QLD + d] = f32_to_bf16(acc[j] + bias);
        }
      } else {
#pragma unroll
        for (int j = 0; j < 8; ++j) {
          int t = mt * 16 + j + (half << 3);
          sVt[(head * HD + d) * VLD + t] = f32_to_bf16(acc[j] + bias);
        }
      }
    }
  }
  __syncthreads();

  // ---- Phase C: per-head attention. 8 heads x 4 M-tiles = 32 row-blocks ----
  for (int i = 0; i < 4; ++i) {
    const int rb = wave * 4 + i;                 // scalar
    const int head = rb >> 2, mt = rb & 3;       // scalar
    const unsigned short* qh = sQ + head * NTOK * QLD;
    const unsigned short* kh = sK + head * NTOK * QLD;
    const unsigned short* vh = sVt + head * HD * VLD;

    Frag aq;
    load_a(qh, QLD, mt * 16, 0, lane, aq);
    v8f s[4];
#pragma unroll
    for (int st = 0; st < 4; ++st) {
      Frag bk;
      load_b(kh, QLD, st * 16, 0, lane, bk);
      v8f z = {0.f, 0.f, 0.f, 0.f, 0.f, 0.f, 0.f, 0.f};
      s[st] = wmma_bf16(aq, bk, z);
    }
    // in-register row softmax: a row lives in one VGPR across a 16-lane group
#pragma unroll
    for (int j = 0; j < 8; ++j) {
#pragma unroll
      for (int st = 0; st < 4; ++st) s[st][j] *= ATTN_SCALE;
      float mx = fmaxf(fmaxf(s[0][j], s[1][j]), fmaxf(s[2][j], s[3][j]));
#pragma unroll
      for (int o = 8; o >= 1; o >>= 1) mx = fmaxf(mx, __shfl_xor(mx, o, 32));
      float sum = 0.f;
#pragma unroll
      for (int st = 0; st < 4; ++st) {
        float e = __expf(s[st][j] - mx);
        s[st][j] = e;
        sum += e;
      }
#pragma unroll
      for (int o = 8; o >= 1; o >>= 1) sum += __shfl_xor(sum, o, 32);
      float inv = 1.0f / sum;
#pragma unroll
      for (int st = 0; st < 4; ++st) s[st][j] *= inv;
    }
    // re-layout probs D-frag -> A-frag via per-wave LDS scratch
    unsigned short* pw = sP + wave * 16 * VLD;   // scalar base
#pragma unroll
    for (int st = 0; st < 4; ++st)
#pragma unroll
      for (int j = 0; j < 8; ++j)
        pw[(j + (half << 3)) * VLD + st * 16 + nloc] = f32_to_bf16(s[st][j]);

    Frag pa0, pa1;
    load_a(pw, VLD, 0, 0,  lane, pa0);
    load_a(pw, VLD, 0, 32, lane, pa1);
#pragma unroll
    for (int dt = 0; dt < 2; ++dt) {
      Frag bv0, bv1;
      load_b(vh, VLD, dt * 16, 0,  lane, bv0);
      load_b(vh, VLD, dt * 16, 32, lane, bv1);
      v8f o = {0.f, 0.f, 0.f, 0.f, 0.f, 0.f, 0.f, 0.f};
      o = wmma_bf16(pa0, bv0, o);
      o = wmma_bf16(pa1, bv1, o);
#pragma unroll
      for (int j = 0; j < 8; ++j)
        sO[(mt * 16 + j + (half << 3)) * XLD + head * 32 + dt * 16 + nloc] =
            f32_to_bf16(o[j]);
    }
  }
  __syncthreads();

  // ---- Phase D: Y = O[64x256] @ Wproj[256x256] + b -> global (f32) ----
  for (int mt = 0; mt < 4; ++mt) {
    Frag a[8];
#pragma unroll
    for (int ks = 0; ks < 8; ++ks) load_a(sO, XLD, mt * 16, ks * 32, lane, a[ks]);
    for (int ii = 0; ii < 2; ++ii) {
      const int nt = wave * 2 + ii;              // scalar
      v8f acc = {0.f, 0.f, 0.f, 0.f, 0.f, 0.f, 0.f, 0.f};
#pragma unroll
      for (int ks = 0; ks < 8; ++ks) {
        Frag bb;
        load_b(wprojT, CDIM, nt * 16, ks * 32, lane, bb);
        acc = wmma_bf16(a[ks], bb, acc);
      }
      int c = nt * 16 + nloc;
      float bias = bproj[c];
#pragma unroll
      for (int j = 0; j < 8; ++j) {
        int t = mt * 16 + j + (half << 3);
        int h = nhi * 8 + (t >> 3), w = nwi * 8 + (t & 7);
        out[(((size_t)(b * 256 + h)) * 256 + w) * 256 + c] = acc[j] + bias;
      }
    }
  }
#undef sO
}

// ---------------------------------------------------------------------------
extern "C" void kernel_launch(void* const* d_in, const int* in_sizes, int n_in,
                              void* d_out, int out_size, void* d_ws, size_t ws_size,
                              hipStream_t stream) {
  const float* x      = (const float*)d_in[0];
  const float* w_qkv  = (const float*)d_in[1];  // [256][768]
  const float* b_qkv  = (const float*)d_in[2];
  const float* w_proj = (const float*)d_in[3];  // [256][256]
  const float* b_proj = (const float*)d_in[4];
  float* out = (float*)d_out;

  unsigned short* wqkvT  = (unsigned short*)d_ws;       // [768][256] bf16
  unsigned short* wprojT = wqkvT + 768 * 256;           // [256][256] bf16

  wa_convert_w<<<(256 * 768 + 255) / 256, 256, 0, stream>>>(w_qkv, wqkvT, 256, 768);
  wa_convert_w<<<(256 * 256 + 255) / 256, 256, 0, stream>>>(w_proj, wprojT, 256, 256);

  wa_window_attn<<<4096, 256, 0, stream>>>(x, wqkvT, b_qkv, wprojT, b_proj, out);
}
